// Quantize_27161373180473
// MI455X (gfx1250) — matrix operand order repaired and checked by reading, not compiled, not run on hardware
//
#include <hip/hip_runtime.h>

typedef __attribute__((ext_vector_type(16))) _Float16 v16h;
typedef __attribute__((ext_vector_type(8)))  float    v8f;

#define N_SAMP   16384        // 8*64*32 samples
#define DIMD     128          // embed dim
#define N_CODE   10000        // codebook entries
#define N_CTILE  625          // code tiles of 16 (10000/16)
#define N_SEG    5            // code-dim split for parallelism
#define TILES_PER_SEG 125     // 625/5
#define N_STILE  1024         // sample tiles of 16 (16384/16)
#define NM_ELEMS 2097152      // 16384*128

// workspace layout (bytes, all 256-aligned)
static constexpr size_t BPACK_OFF = 0;                    // 10000*128 f16 = 2,560,000
static constexpr size_t HN_OFF    = 2560000;              // 10000 f32    =    40,000
static constexpr size_t CT_OFF    = 2600192;              // 10000*128 f32 = 5,120,000
static constexpr size_t SEGV_OFF  = 7720192;              // 5*16384 f32  =   327,680
static constexpr size_t SEGI_OFF  = 8047872;              // 5*16384 i32  =   327,680
static constexpr size_t IDX_OFF   = 8375552;              // 16384 i32    =    65,536
static constexpr size_t PART_OFF  = 8441088;              // 8192 f32     =    32,768

__device__ __forceinline__ v8f wmma_f16(v16h a, v16h b, v8f c) {
  // D = A(16x32,f16) x B(32x16,f16) + C(16x16,f32)
  return __builtin_amdgcn_wmma_f32_16x16x32_f16(
      /*neg_a=*/false, a, /*neg_b=*/false, b,
      /*c_mod=*/(short)0, c, /*reuse_a=*/false, /*reuse_b=*/false);
}

// ---------------------------------------------------------------------------
// Pack codebook into WMMA-B fragment layout, half-norms, and transposed f32
// copy for the final gather. B layout (16-bit, 32x16 chunk): lane L holds column
// N=L%16 of the chunk; its elements 0..7 are K=8*(L/16)+0..7, 8..15 are K=16+8*(L/16)+0..7.
// ---------------------------------------------------------------------------
__global__ void __launch_bounds__(256) vq_prep(const float* __restrict__ C,
                                               _Float16* __restrict__ bpack,
                                               float* __restrict__ hnorm,
                                               float* __restrict__ CT) {
  int j = blockIdx.x * 256 + threadIdx.x;
  if (j >= N_CODE) return;
  const int t = j >> 4, n = j & 15;
  float acc = 0.f;
  for (int d = 0; d < DIMD; ++d) {
    float c = C[d * N_CODE + j];             // coalesced across j
    acc += c * c;
    CT[j * DIMD + d] = c;
    const int kc = d >> 5, kk = d & 31, lane = n + 16 * ((kk >> 3) & 1);
    bpack[(size_t)((t * 4 + kc) * 32 + lane) * 16 + (kk & 7) + ((kk >> 4) << 3)] = (_Float16)c;
  }
  hnorm[j] = 0.5f * acc;                     // score = 0.5||c||^2 - x.c
}

// ---------------------------------------------------------------------------
// Fused GEMM + argmin. One wave per (sample tile, code segment).
// ---------------------------------------------------------------------------
__global__ void __launch_bounds__(32) vq_argmin(const float* __restrict__ X,
                                                const _Float16* __restrict__ bpackh,
                                                const float* __restrict__ hnorm,
                                                float* __restrict__ segval,
                                                int* __restrict__ segidx) {
  const int tile = blockIdx.x;               // sample tile (0..1023)
  const int seg  = blockIdx.y;               // code segment (0..4)
  const int lane = threadIdx.x & 31;
  const int col  = lane & 15;
  const int half = lane >> 4;

  // A fragment (16-bit A 16x32): lane holds row M=lane&15.
  // half 0 -> K {kc*32+0..7, kc*32+16..23}; half 1 -> K {+8..15, +24..31}
  const float* __restrict__ arow = X + (size_t)(tile * 16 + col) * DIMD;
  v16h a[4];
#pragma unroll
  for (int kc = 0; kc < 4; ++kc) {
    const int base = kc * 32 + half * 8;
#pragma unroll
    for (int e = 0; e < 8; ++e) {
      a[kc][e]     = (_Float16)arow[base + e];
      a[kc][8 + e] = (_Float16)arow[base + 16 + e];
    }
  }

  const v16h* __restrict__ bp = (const v16h*)bpackh;
  float bestv[8];
  int   besti[8];
#pragma unroll
  for (int m = 0; m < 8; ++m) { bestv[m] = 3.0e38f; besti[m] = 0; }

  const int t0 = seg * TILES_PER_SEG;
  for (int t = t0; t < t0 + TILES_PER_SEG; ++t) {
    const int tb = t * 128 + lane;           // v16h index into packed B
    v16h b0 = bp[tb];
    v16h b1 = bp[tb + 32];
    v16h b2 = bp[tb + 64];
    v16h b3 = bp[tb + 96];
    const float hn = hnorm[t * 16 + col];
    v8f acc0 = {};
    v8f acc1 = {};
    acc0 = wmma_f16(a[0], b0, acc0);         // two independent chains for ILP
    acc1 = wmma_f16(a[1], b1, acc1);
    acc0 = wmma_f16(a[2], b2, acc0);
    acc1 = wmma_f16(a[3], b3, acc1);
    const int j = t * 16 + col;
#pragma unroll
    for (int m = 0; m < 8; ++m) {
      // D layout: element m of lane L -> (M = m + 8*half, N = col)
      const float score = hn - (acc0[m] + acc1[m]);
      if (score < bestv[m]) { bestv[m] = score; besti[m] = j; }
    }
  }

  // min+argmin butterfly across the 16 lanes of each half (bits 0..3 only,
  // so reduction stays within rows {0..7}+8*half). Ties -> smaller index.
#pragma unroll
  for (int off = 8; off >= 1; off >>= 1) {
#pragma unroll
    for (int m = 0; m < 8; ++m) {
      const float ov = __shfl_xor(bestv[m], off, 32);
      const int   oi = __shfl_xor(besti[m], off, 32);
      if (ov < bestv[m] || (ov == bestv[m] && oi < besti[m])) {
        bestv[m] = ov; besti[m] = oi;
      }
    }
  }
  if (col == 0) {
#pragma unroll
    for (int m = 0; m < 8; ++m) {
      const int r = tile * 16 + half * 8 + m;
      segval[seg * N_SAMP + r] = bestv[m];
      segidx[seg * N_SAMP + r] = besti[m];
    }
  }
}

// ---------------------------------------------------------------------------
// Combine per-segment winners (segments are ascending-j, strict < keeps the
// first index on ties, matching jnp.argmin).
// ---------------------------------------------------------------------------
__global__ void __launch_bounds__(256) vq_combine(const float* __restrict__ segval,
                                                  const int* __restrict__ segidx,
                                                  int* __restrict__ argidx) {
  const int i = blockIdx.x * 256 + threadIdx.x;
  if (i >= N_SAMP) return;
  float bv = 3.0e38f;
  int   bi = 0;
  for (int s = 0; s < N_SEG; ++s) {
    const float v = segval[s * N_SAMP + i];
    const int   j = segidx[s * N_SAMP + i];
    if (v < bv) { bv = v; bi = j; }
  }
  argidx[i] = bi;
}

// ---------------------------------------------------------------------------
// Gather outputs (exact f32 codebook values) + per-block partial loss sums.
// ---------------------------------------------------------------------------
__global__ void __launch_bounds__(256) vq_gather(const float* __restrict__ X,
                                                 const float* __restrict__ CT,
                                                 const int* __restrict__ argidx,
                                                 float* __restrict__ out,
                                                 float* __restrict__ partials) {
  const int gid = blockIdx.x * 256 + threadIdx.x;
  const int i = gid >> 7, d = gid & 127;
  const int idx = argidx[i];
  const float q = CT[(size_t)idx * DIMD + d];  // coalesced 512B per sample
  const float x = X[gid];
  out[gid] = q;                                // outputs == quantize (STE)
  const float diff = q - x;
  float s = diff * diff;
#pragma unroll
  for (int off = 16; off >= 1; off >>= 1) s += __shfl_xor(s, off, 32);
  __shared__ float red[8];
  if ((threadIdx.x & 31) == 0) red[threadIdx.x >> 5] = s;
  __syncthreads();
  if (threadIdx.x == 0) {
    float tot = 0.f;
#pragma unroll
    for (int w = 0; w < 8; ++w) tot += red[w];
    partials[blockIdx.x] = tot;
  }
}

__global__ void __launch_bounds__(256) vq_loss(const float* __restrict__ partials,
                                               float* __restrict__ loss) {
  float s = 0.f;
  for (int i = threadIdx.x; i < 8192; i += 256) s += partials[i];
#pragma unroll
  for (int off = 16; off >= 1; off >>= 1) s += __shfl_xor(s, off, 32);
  __shared__ float red[8];
  if ((threadIdx.x & 31) == 0) red[threadIdx.x >> 5] = s;
  __syncthreads();
  if (threadIdx.x == 0) {
    float tot = 0.f;
    for (int w = 0; w < 8; ++w) tot += red[w];
    loss[0] = 1.25f * tot / (float)NM_ELEMS;   // q_loss + 0.25*e_loss
  }
}

extern "C" void kernel_launch(void* const* d_in, const int* in_sizes, int n_in,
                              void* d_out, int out_size, void* d_ws, size_t ws_size,
                              hipStream_t stream) {
  const float* X = (const float*)d_in[0];   // (16384, 128) f32
  const float* C = (const float*)d_in[1];   // (128, 10000) f32
  float* out = (float*)d_out;               // 16384*128 outputs + 1 loss

  char* ws = (char*)d_ws;
  _Float16* bpack   = (_Float16*)(ws + BPACK_OFF);
  float*    hnorm   = (float*)(ws + HN_OFF);
  float*    CT      = (float*)(ws + CT_OFF);
  float*    segval  = (float*)(ws + SEGV_OFF);
  int*      segidx  = (int*)(ws + SEGI_OFF);
  int*      argidx  = (int*)(ws + IDX_OFF);
  float*    partials= (float*)(ws + PART_OFF);

  vq_prep<<<(N_CODE + 255) / 256, 256, 0, stream>>>(C, bpack, hnorm, CT);
  vq_argmin<<<dim3(N_STILE, N_SEG), 32, 0, stream>>>(X, bpack, hnorm, segval, segidx);
  vq_combine<<<N_SAMP / 256, 256, 0, stream>>>(segval, segidx, argidx);
  vq_gather<<<NM_ELEMS / 256, 256, 0, stream>>>(X, CT, argidx, out, partials);
  vq_loss<<<1, 256, 0, stream>>>(partials, out + NM_ELEMS);
}
